// KimiSparseMoeBlock_74371653698289
// MI455X (gfx1250) — matrix operand order repaired and checked by reading, not compiled, not run on hardware
//
#include <hip/hip_runtime.h>
#include <hip/hip_bf16.h>

// ---------------- problem constants ----------------
#define T   4096
#define D   1024
#define E   8
#define F   512
#define SH  1024

typedef __bf16 bf16;
typedef __attribute__((ext_vector_type(16))) __bf16 v16bf;
typedef __attribute__((ext_vector_type(8)))  __bf16 v8bf;
typedef __attribute__((ext_vector_type(4)))  __bf16 v4bf;
typedef __attribute__((ext_vector_type(8)))  float  v8f;
typedef __attribute__((ext_vector_type(4)))  float  f32x4;
typedef __attribute__((ext_vector_type(4)))  unsigned int u32x4;

// LDS tile strides (elements). 32 K-elements + 8 pad = 80B rows: keeps every
// fragment chunk 16B-aligned for ds_load_b128 while breaking bank alignment.
#define LDA 40
#define LDB 40

__device__ __forceinline__ float fast_sigmoid(float x) {
    return __builtin_amdgcn_rcpf(1.f + __expf(-x));   // v_rcp_f32, no IEEE div
}

// Async DMA: per-lane 16B global -> LDS, tracked by ASYNCcnt (ISA 15.18.3).
__device__ __forceinline__ void async_b128_to_lds(const void* gsrc, void* ldst) {
    const unsigned dst = (unsigned)(unsigned long long)ldst;   // LDS = addr[31:0]
    asm volatile("global_load_async_to_lds_b128 %0, %1, off"
                 :: "v"(dst), "v"(gsrc) : "memory");
}
__device__ __forceinline__ void wait_async0() {
    asm volatile("s_wait_asynccnt 0x0" ::: "memory");
}

// ---------------- WMMA fragment loads (per CDNA5 ISA 7.12.2 layouts) --------
// A (16x32 bf16): lanes 0-15 hold M=lane; VGPR0-3 = K[kh*8 .. kh*8+7],
// VGPR4-7 = K[16+kh*8 ..], where kh = lane>>4.
__device__ __forceinline__ v16bf load_a_frag(const bf16* tile, int lane) {
    const int m  = lane & 15;
    const int kh = (lane >> 4) * 8;
    const bf16* p = tile + m * LDA;
    v8bf lo = *(const v8bf*)(p + kh);
    v8bf hi = *(const v8bf*)(p + 16 + kh);
    v16bf r;
#pragma unroll
    for (int i = 0; i < 8; ++i) { r[i] = lo[i]; r[i + 8] = hi[i]; }
    return r;
}

// B (32x16 bf16): lane = N column; elements 0-7 hold K[kh*8 .. kh*8+7],
// elements 8-15 hold K[16+kh*8 ..], kh = lane>>4. LDS tile stored [n][k] (k contiguous).
__device__ __forceinline__ v16bf load_b_frag(const bf16* tile, int lane) {
    const int n  = lane & 15;
    const int kh = (lane >> 4) * 8;
    const bf16* p = tile + n * LDB + kh;
    v8bf lo = *(const v8bf*)(p);
    v8bf hi = *(const v8bf*)(p + 16);
    v16bf r;
#pragma unroll
    for (int i = 0; i < 8; ++i) { r[i] = lo[i]; r[i + 8] = hi[i]; }
    return r;
}

// ---------------- Kernel 1: router gate ------------------------------------
__global__ void __launch_bounds__(256) gate_kernel(
    const float* __restrict__ x, const float* __restrict__ gw,
    const float* __restrict__ gb, float* __restrict__ ew)
{
    const int token = blockIdx.x * 8 + (threadIdx.x >> 5);
    const int lane  = threadIdx.x & 31;
    const float* xr = x + (long)token * D;

    float acc[E];
#pragma unroll
    for (int e = 0; e < E; ++e) acc[e] = 0.f;

    for (int j = 0; j < D / 32; ++j) {
        const float xv = xr[j * 32 + lane];
#pragma unroll
        for (int e = 0; e < E; ++e)
            acc[e] += xv * gw[e * D + j * 32 + lane];
    }
#pragma unroll
    for (int e = 0; e < E; ++e)
#pragma unroll
        for (int off = 16; off > 0; off >>= 1)
            acc[e] += __shfl_xor(acc[e], off, 32);

    float sc[E], scb[E];
#pragma unroll
    for (int e = 0; e < E; ++e) {
        sc[e]  = fast_sigmoid(acc[e]);            // unbiased scores -> weights
        scb[e] = sc[e] + gb[e];                   // biased scores -> selection
    }
    float gs[4];
#pragma unroll
    for (int g = 0; g < 4; ++g) gs[g] = scb[2 * g] + scb[2 * g + 1];
    int g1 = 0;
#pragma unroll
    for (int g = 1; g < 4; ++g) if (gs[g] > gs[g1]) g1 = g;
    int g2 = (g1 == 0) ? 1 : 0;
#pragma unroll
    for (int g = 0; g < 4; ++g) if (g != g1 && gs[g] > gs[g2]) g2 = g;
    float tmp[E];
#pragma unroll
    for (int e = 0; e < E; ++e) {
        const int g = e >> 1;
        tmp[e] = (g == g1 || g == g2) ? scb[e] : 0.f;
    }
    int i1 = 0;
#pragma unroll
    for (int e = 1; e < E; ++e) if (tmp[e] > tmp[i1]) i1 = e;
    int i2 = (i1 == 0) ? 1 : 0;
#pragma unroll
    for (int e = 0; e < E; ++e) if (e != i1 && tmp[e] > tmp[i2]) i2 = e;

    const float wa = sc[i1], wb = sc[i2];
    const float inv = 2.5f * __builtin_amdgcn_rcpf(wa + wb + 1e-20f);
    if (lane < E) {
        float v = (lane == i1) ? wa * inv : (lane == i2) ? wb * inv : 0.f;
        ew[(long)token * E + lane] = v;
    }
}

// ---------------- Kernel 2: x f32 -> bf16 -----------------------------------
__global__ void __launch_bounds__(256) cvt_kernel(
    const float* __restrict__ x, bf16* __restrict__ xb)
{
    const long i = ((long)blockIdx.x * 256 + threadIdx.x) * 4;
    f32x4 v = *(const f32x4*)(x + i);
    v4bf b;
#pragma unroll
    for (int k = 0; k < 4; ++k) b[k] = (bf16)v[k];
    *(v4bf*)(xb + i) = b;
}

// ---------------- fetch / commit halves of tile staging ---------------------
// A tile via async DMA (raw bf16 copy; no VGPR staging, writes the ping-pong
// *other* buffer while WMMAs read the current one).
__device__ __forceinline__ void async_A(const bf16* A, long lda, int m0, int kb,
                                        int t, bf16* ldsbuf) {
#pragma unroll
    for (int p = 0; p < 2; ++p) {
        const int row = (t >> 2) + 64 * p;
        const int kp  = (t & 3) * 8;
        async_b128_to_lds(A + (long)(m0 + row) * lda + kb + kp,
                          ldsbuf + row * LDA + kp);
    }
}

// A tile, bf16 with optional per-token-row scale (combine weight folded in)
__device__ __forceinline__ void fetch_A_scaled(const bf16* A, long lda,
                                               const float* sc, int m0, int kb,
                                               int t, v8bf r[2], float s[2]) {
#pragma unroll
    for (int p = 0; p < 2; ++p) {
        const int row = (t >> 2) + 64 * p;
        const int kp  = (t & 3) * 8;
        const int gr  = m0 + row;
        r[p] = *(const v8bf*)(A + (long)gr * lda + kb + kp);
        s[p] = sc ? sc[(long)gr * E] : 1.f;
    }
}
__device__ __forceinline__ void commit_A_scaled(bf16* lds, int t,
                                                const v8bf r[2], const float s[2]) {
#pragma unroll
    for (int p = 0; p < 2; ++p) {
        const int row = (t >> 2) + 64 * p;
        const int kp  = (t & 3) * 8;
        v8bf v;
#pragma unroll
        for (int i = 0; i < 8; ++i) v[i] = (bf16)((float)r[p][i] * s[p]);
        *(v8bf*)&lds[row * LDA + kp] = v;
    }
}

// B tile: f32 weights -> bf16 (register path: needs conversion)
__device__ __forceinline__ void fetch_B(const float* B, long ldb, int n0,
                                        int kb, int t, f32x4 r[2]) {
#pragma unroll
    for (int p = 0; p < 2; ++p) {
        const int row = (t >> 3) + 32 * p;
        const int kp  = (t & 7) * 4;
        r[p] = *(const f32x4*)(B + (long)(n0 + row) * ldb + kb + kp);
    }
}
__device__ __forceinline__ void commit_B(bf16* lds, int t, const f32x4 r[2]) {
#pragma unroll
    for (int p = 0; p < 2; ++p) {
        const int row = (t >> 3) + 32 * p;
        const int kp  = (t & 7) * 4;
        v4bf b;
#pragma unroll
        for (int i = 0; i < 4; ++i) b[i] = (bf16)r[p][i];
        *(v4bf*)&lds[row * LDB + kp] = b;
    }
}

// ---------------- WMMA inner step -------------------------------------------
__device__ __forceinline__ void mma_step(const bf16* ldsA, const bf16* ldsB,
                                         int wm, int wn, int lane, v8f acc[2][2])
{
    v16bf a[2], b[2];
#pragma unroll
    for (int tm = 0; tm < 2; ++tm)
        a[tm] = load_a_frag(ldsA + (wm + 16 * tm) * LDA, lane);
#pragma unroll
    for (int tn = 0; tn < 2; ++tn)
        b[tn] = load_b_frag(ldsB + (wn + 16 * tn) * LDB, lane);
#pragma unroll
    for (int tm = 0; tm < 2; ++tm)
#pragma unroll
        for (int tn = 0; tn < 2; ++tn)
            acc[tm][tn] = __builtin_amdgcn_wmma_f32_16x16x32_bf16(
                false, a[tm], false, b[tn], (short)0, acc[tm][tn], false, false);
}

// ---------------- Kernel 3/4: fused SwiGLU up-projection --------------------
// H[t,n] = silu(x . Wg[n]) * (x . Wu[n]).  Block tile 128x64, 8 waves of
// 32x32, BK=32, double-buffered LDS; A via async DMA, B via register path.
__global__ void __launch_bounds__(256) up_kernel(
    const bf16* __restrict__ xb, const float* __restrict__ WgB,
    const float* __restrict__ WuB, bf16* __restrict__ HB,
    int N, long wstride, long ostride)
{
    __shared__ __align__(16) bf16 ldsA [2][128 * LDA];
    __shared__ __align__(16) bf16 ldsBg[2][ 64 * LDB];
    __shared__ __align__(16) bf16 ldsBu[2][ 64 * LDB];

    const int e = blockIdx.z;
    const float* Wg = WgB + (long)e * wstride;
    const float* Wu = WuB + (long)e * wstride;
    bf16* Hout = HB + (long)e * ostride;

    const int m0 = blockIdx.x * 128, n0 = blockIdx.y * 64;
    const int t = threadIdx.x, lane = t & 31, wid = t >> 5;
    const int wm = (wid & 3) * 32, wn = (wid >> 2) * 32;

    v8f accg[2][2], accu[2][2];
#pragma unroll
    for (int i = 0; i < 2; ++i)
#pragma unroll
        for (int j = 0; j < 2; ++j)
#pragma unroll
            for (int k = 0; k < 8; ++k) { accg[i][j][k] = 0.f; accu[i][j][k] = 0.f; }

    f32x4 fg[2], fu[2];
    // prologue: stage tile 0
    async_A(xb, D, m0, 0, t, ldsA[0]);
    fetch_B(Wg, D, n0, 0, t, fg);
    fetch_B(Wu, D, n0, 0, t, fu);
    commit_B(ldsBg[0], t, fg);
    commit_B(ldsBu[0], t, fu);
    wait_async0();
    __syncthreads();

    const int NK = D / 32;
    for (int s = 0; s < NK; ++s) {
        const int cur = s & 1;
        const bool more = (s + 1 < NK);
        if (more) {                           // next-tile: DMA A, load B
            const int kb = (s + 1) * 32;
            async_A(xb, D, m0, kb, t, ldsA[cur ^ 1]);
            fetch_B(Wg, D, n0, kb, t, fg);
            fetch_B(Wu, D, n0, kb, t, fu);
        }
        // math on current buffer (covers DMA/fetch latency)
        v16bf a[2], bg[2], bu[2];
#pragma unroll
        for (int tm = 0; tm < 2; ++tm)
            a[tm] = load_a_frag(ldsA[cur] + (wm + 16 * tm) * LDA, lane);
#pragma unroll
        for (int tn = 0; tn < 2; ++tn) {
            bg[tn] = load_b_frag(ldsBg[cur] + (wn + 16 * tn) * LDB, lane);
            bu[tn] = load_b_frag(ldsBu[cur] + (wn + 16 * tn) * LDB, lane);
        }
#pragma unroll
        for (int tm = 0; tm < 2; ++tm)
#pragma unroll
            for (int tn = 0; tn < 2; ++tn) {
                accg[tm][tn] = __builtin_amdgcn_wmma_f32_16x16x32_bf16(
                    false, a[tm], false, bg[tn], (short)0, accg[tm][tn], false, false);
                accu[tm][tn] = __builtin_amdgcn_wmma_f32_16x16x32_bf16(
                    false, a[tm], false, bu[tn], (short)0, accu[tm][tn], false, false);
            }
        if (more) {
            commit_B(ldsBg[cur ^ 1], t, fg);
            commit_B(ldsBu[cur ^ 1], t, fu);
            wait_async0();                    // A DMA landed in other buffer
        }
        __syncthreads();
    }

    // SiLU epilogue. C layout: VGPR i -> M = i + (lane>>4)*8, N = lane&15.
#pragma unroll
    for (int tm = 0; tm < 2; ++tm)
#pragma unroll
        for (int tn = 0; tn < 2; ++tn)
#pragma unroll
            for (int i = 0; i < 8; ++i) {
                const float g = accg[tm][tn][i];
                const float u = accu[tm][tn][i];
                const float h = g * fast_sigmoid(g) * u;
                const int r = m0 + wm + 16 * tm + (lane >> 4) * 8 + i;
                const int c = n0 + wn + 16 * tn + (lane & 15);
                Hout[(long)r * N + c] = (bf16)h;
            }
}

// ---------------- Kernel 5: down-projection + combine -----------------------
// out[t,d] = sum_e ew[t,e]*(H_e @ W2_e^T)[t,d] + (S @ Wd^T)[t,d]
// Flattened K = 8*F + SH = 5120 into one 160-step pipelined loop; ew folded
// into A rows at commit time (scale=1 for the shared-expert phase).
__device__ __forceinline__ void resolve_step(
    int step, const bf16* H, const float* w2, const bf16* S, const float* Wd,
    const float* ew, const bf16*& A, long& lda, const float*& sc,
    const float*& B, long& ldb, int& kb)
{
    const int RSTEPS = E * (F / 32);          // 128 routed steps
    if (step < RSTEPS) {
        const int e = step >> 4;              // F/32 == 16
        kb  = (step & 15) * 32;
        A   = H + (long)e * T * F;  lda = F;  sc = ew + e;
        B   = w2 + (long)e * D * F; ldb = F;
    } else {
        kb  = (step - RSTEPS) * 32;
        A   = S;  lda = SH;  sc = nullptr;
        B   = Wd; ldb = SH;
    }
}

__global__ void __launch_bounds__(256) down_kernel(
    const bf16* __restrict__ H, const float* __restrict__ w2,
    const bf16* __restrict__ S, const float* __restrict__ Wd,
    const float* __restrict__ ew, float* __restrict__ out)
{
    __shared__ __align__(16) bf16 ldsA[2][128 * LDA];
    __shared__ __align__(16) bf16 ldsB[2][ 64 * LDB];

    const int m0 = blockIdx.x * 128, n0 = blockIdx.y * 64;
    const int t = threadIdx.x, lane = t & 31, wid = t >> 5;
    const int wm = (wid & 3) * 32, wn = (wid >> 2) * 32;

    v8f acc[2][2];
#pragma unroll
    for (int i = 0; i < 2; ++i)
#pragma unroll
        for (int j = 0; j < 2; ++j)
#pragma unroll
            for (int k = 0; k < 8; ++k) acc[i][j][k] = 0.f;

    const int NSTEP = E * (F / 32) + SH / 32; // 160
    const bf16* A; const float* B; const float* sc; long lda, ldb; int kb;
    v8bf fa[2]; float fs[2]; f32x4 fb[2];

    resolve_step(0, H, w2, S, Wd, ew, A, lda, sc, B, ldb, kb);
    fetch_A_scaled(A, lda, sc, m0, kb, t, fa, fs);
    fetch_B(B, ldb, n0, kb, t, fb);
    commit_A_scaled(ldsA[0], t, fa, fs);
    commit_B(ldsB[0], t, fb);
    __syncthreads();

    for (int s = 0; s < NSTEP; ++s) {
        const int cur = s & 1;
        const bool more = (s + 1 < NSTEP);
        if (more) {
            resolve_step(s + 1, H, w2, S, Wd, ew, A, lda, sc, B, ldb, kb);
            fetch_A_scaled(A, lda, sc, m0, kb, t, fa, fs);
            fetch_B(B, ldb, n0, kb, t, fb);
        }
        mma_step(ldsA[cur], ldsB[cur], wm, wn, lane, acc);
        if (more) {
            commit_A_scaled(ldsA[cur ^ 1], t, fa, fs);
            commit_B(ldsB[cur ^ 1], t, fb);
        }
        __syncthreads();
    }

#pragma unroll
    for (int tm = 0; tm < 2; ++tm)
#pragma unroll
        for (int tn = 0; tn < 2; ++tn)
#pragma unroll
            for (int i = 0; i < 8; ++i) {
                const int r = m0 + wm + 16 * tm + (lane >> 4) * 8 + i;
                const int c = n0 + wn + 16 * tn + (lane & 15);
                out[(long)r * D + c] = acc[tm][tn][i];
            }
}

// ---------------- launch -----------------------------------------------------
extern "C" void kernel_launch(void* const* d_in, const int* in_sizes, int n_in,
                              void* d_out, int out_size, void* d_ws, size_t ws_size,
                              hipStream_t stream) {
    const float* x    = (const float*)d_in[0];
    const float* gw   = (const float*)d_in[1];
    const float* gb   = (const float*)d_in[2];
    const float* w1   = (const float*)d_in[3];
    const float* w2   = (const float*)d_in[4];
    const float* w3   = (const float*)d_in[5];
    const float* sgw  = (const float*)d_in[6];
    const float* suw  = (const float*)d_in[7];
    const float* sdw  = (const float*)d_in[8];
    float* out = (float*)d_out;

    char* ws = (char*)d_ws;
    const size_t OFF_XB = 131072;                             // after ew (T*E*4)
    const size_t OFF_H  = OFF_XB + (size_t)T * D * 2;
    const size_t OFF_S  = OFF_H  + (size_t)E * T * F * 2;
    float* ew = (float*)(ws);
    bf16*  xb = (bf16*) (ws + OFF_XB);
    bf16*  Hb = (bf16*) (ws + OFF_H);
    bf16*  Sb = (bf16*) (ws + OFF_S);

    gate_kernel<<<T / 8, 256, 0, stream>>>(x, gw, gb, ew);
    cvt_kernel <<<(T * (long)D) / (256 * 4), 256, 0, stream>>>(x, xb);
    up_kernel<<<dim3(T / 128, F / 64, E), 256, 0, stream>>>(
        xb, w1, w3, Hb, F, (long)F * D, (long)T * F);
    up_kernel<<<dim3(T / 128, SH / 64, 1), 256, 0, stream>>>(
        xb, sgw, suw, Sb, SH, 0L, 0L);
    down_kernel<<<dim3(T / 128, D / 64), 256, 0, stream>>>(
        Hb, w2, Sb, sdw, ew, out);
    (void)in_sizes; (void)n_in; (void)out_size; (void)ws_size;
}
